// RingAttention_1099511627835
// MI455X (gfx1250) — matrix operand order repaired and checked by reading, not compiled, not run on hardware
//
#include <hip/hip_runtime.h>
#include <math.h>

typedef __attribute__((ext_vector_type(16))) __bf16 v16bf;
typedef __attribute__((ext_vector_type(8)))  __bf16 bf16x8;
typedef __attribute__((ext_vector_type(8)))  float  v8f;

static constexpr int S_LEN  = 4096;
static constexpr int DMODEL = 2048;
static constexpr int NH     = 16;
static constexpr int NKV    = 4;
static constexpr int HD     = 128;

__device__ __forceinline__ v16bf cat8(bf16x8 lo, bf16x8 hi) {
    return __builtin_shufflevector(lo, hi, 0,1,2,3,4,5,6,7,8,9,10,11,12,13,14,15);
}

__device__ __forceinline__ v8f wmma_bf16(v16bf a, v16bf b, v8f c) {
    return __builtin_amdgcn_wmma_f32_16x16x32_bf16(false, a, false, b, (short)0, c, false, false);
}

// CDNA5 async memory->LDS copy (ASYNCcnt tracked, no VGPR round trip).
// ldsaddr: LDS byte address (low 32 bits of generic shared pointer, ISA 10.2).
__device__ __forceinline__ void async_copy_b128(void* lds, const void* gptr) {
    unsigned la = (unsigned)(unsigned long long)lds;
    asm volatile("global_load_async_to_lds_b128 %0, %1, off"
                 :: "v"(la), "v"(gptr) : "memory");
}
__device__ __forceinline__ void async_wait0() {
    asm volatile("s_wait_asynccnt 0x0" ::: "memory");
}

// xor-shuffle within each 16-lane half, on the VALU (v_permlane16_b32) instead
// of the LDS pipe (ds_bpermute) so it doesn't contend with WMMA operand loads.
// S0/S1 pack 8 x 4-bit lane selects each: lane i reads lane i^MASK.
template <int MASK, unsigned S0, unsigned S1>
__device__ __forceinline__ float lanexor16(float x) {
#if __has_builtin(__builtin_amdgcn_permlane16)
    int xi = __float_as_int(x);
    return __int_as_float(
        __builtin_amdgcn_permlane16(xi, xi, (int)S0, (int)S1, false, false));
#else
    return __shfl_xor(x, MASK, 32);
#endif
}
__device__ __forceinline__ float red_max16(float v) {
    v = fmaxf(v, lanexor16<1, 0x67452301u, 0xEFCDAB89u>(v));
    v = fmaxf(v, lanexor16<2, 0x54761032u, 0xDCFE98BAu>(v));
    v = fmaxf(v, lanexor16<4, 0x32107654u, 0xBA98FEDCu>(v));
    v = fmaxf(v, lanexor16<8, 0xFEDCBA98u, 0x76543210u>(v));
    return v;
}
__device__ __forceinline__ float red_sum16(float v) {
    v += lanexor16<1, 0x67452301u, 0xEFCDAB89u>(v);
    v += lanexor16<2, 0x54761032u, 0xDCFE98BAu>(v);
    v += lanexor16<4, 0x32107654u, 0xBA98FEDCu>(v);
    v += lanexor16<8, 0xFEDCBA98u, 0x76543210u>(v);
    return v;
}

// ---------------------------------------------------------------------------
// f32 -> bf16 elementwise convert (row-major A operands)
// ---------------------------------------------------------------------------
__global__ void convert_bf16_kernel(const float* __restrict__ x,
                                    __bf16* __restrict__ y, int n) {
    int i = blockIdx.x * 256 + threadIdx.x;
    if (i < n) y[i] = (__bf16)x[i];
}

// ---------------------------------------------------------------------------
// W (K x N, f32) -> Wt (N x K, bf16): B operand wants N-major rows of K.
// ---------------------------------------------------------------------------
__global__ void transpose_bf16_kernel(const float* __restrict__ w,
                                      __bf16* __restrict__ wt, int K, int N) {
    int i = blockIdx.x * 256 + threadIdx.x;
    if (i >= K * N) return;
    int n = i / K;
    int k = i - n * K;
    wt[i] = (__bf16)w[(size_t)k * N + n];
}

// ---------------------------------------------------------------------------
// Tiled bf16 WMMA GEMM: C(MxN,f32) = A(MxK) * Bt(NxK)^T
// Block tile 128x128, K step 32, 256 threads (8 waves).
// Wave tile 32x64 (2 M x 4 N subtiles = 8 WMMA / K-step).
// Double-buffered LDS fed by GLOBAL_LOAD_ASYNC_TO_LDS, 1 barrier per K-step.
// ---------------------------------------------------------------------------
template <int M, int N, int K>
__global__ __launch_bounds__(256) void gemm_bf16_kernel(
    const __bf16* __restrict__ A, const __bf16* __restrict__ Bt,
    float* __restrict__ C) {
    __shared__ __attribute__((aligned(16))) __bf16 As[2][128][40];
    __shared__ __attribute__((aligned(16))) __bf16 Bs[2][128][40];

    const int tid  = threadIdx.x;
    const int wave = tid >> 5;
    const int lane = tid & 31;
    const int half = lane >> 4;
    const int l16  = lane & 15;
    const int m0 = blockIdx.y * 128;
    const int n0 = blockIdx.x * 128;
    const int wm = (wave >> 1) * 32;
    const int wn = (wave & 1) * 64;

    auto stage = [&](int kt, int buf) {
        #pragma unroll
        for (int c = 0; c < 2; ++c) {
            int chunk = tid * 2 + c;
            int row = chunk >> 2;
            int col = (chunk & 3) * 8;
            async_copy_b128(&As[buf][row][col],
                            &A[(size_t)(m0 + row) * K + kt * 32 + col]);
            async_copy_b128(&Bs[buf][row][col],
                            &Bt[(size_t)(n0 + row) * K + kt * 32 + col]);
        }
    };

    v8f acc[2][4] = {};
    constexpr int nk = K / 32;

    stage(0, 0);
    async_wait0();
    __syncthreads();

    for (int kt = 0; kt < nk; ++kt) {
        const int buf = kt & 1;
        if (kt + 1 < nk) stage(kt + 1, buf ^ 1);   // overlaps the WMMAs below

        v16bf av[2];
        #pragma unroll
        for (int mt = 0; mt < 2; ++mt) {
            const __bf16* ap = &As[buf][wm + mt * 16 + l16][0];
            av[mt] = cat8(*(const bf16x8*)&ap[half * 8],
                          *(const bf16x8*)&ap[16 + half * 8]);
        }
        #pragma unroll
        for (int nt = 0; nt < 4; ++nt) {
            const __bf16* bp = &Bs[buf][wn + nt * 16 + l16][0];
            v16bf bv = cat8(*(const bf16x8*)&bp[half * 8], *(const bf16x8*)&bp[16 + half * 8]);
            #pragma unroll
            for (int mt = 0; mt < 2; ++mt)
                acc[mt][nt] = wmma_bf16(av[mt], bv, acc[mt][nt]);
        }
        if (kt + 1 < nk) async_wait0();
        __syncthreads();
    }

    #pragma unroll
    for (int mt = 0; mt < 2; ++mt)
        #pragma unroll
        for (int nt = 0; nt < 4; ++nt)
            #pragma unroll
            for (int r = 0; r < 8; ++r) {
                int row = m0 + wm + mt * 16 + half * 8 + r;  // C layout: M = r + 8*half
                int col = n0 + wn + nt * 16 + l16;
                C[(size_t)row * N + col] = acc[mt][nt][r];
            }
}

// ---------------------------------------------------------------------------
// RoPE + repack: X[S, nh*HD] f32 -> Y[nh, S, HD] bf16 (scale folded for Q)
// ---------------------------------------------------------------------------
__global__ void rope_pack_kernel(const float* __restrict__ X,
                                 __bf16* __restrict__ Y,
                                 int nheads, float scale) {
    int i = blockIdx.x * 256 + threadIdx.x;
    int total = S_LEN * nheads * (HD / 2);
    if (i >= total) return;
    int d = i % (HD / 2);
    int t = i / (HD / 2);
    int h = t % nheads;
    int s = t / nheads;
    size_t base = (size_t)s * nheads * HD + (size_t)h * HD;
    float xr = X[base + d];
    float xi = X[base + 64 + d];
    float ang = (float)s * __powf(10000.0f, -(float)d * (1.0f / 64.0f));
    float c = __cosf(ang), sn = __sinf(ang);
    size_t o = (size_t)h * S_LEN * HD + (size_t)s * HD + d;
    Y[o]      = (__bf16)((xr * c - xi * sn) * scale);
    Y[o + 64] = (__bf16)((xr * sn + xi * c) * scale);
}

// ---------------------------------------------------------------------------
// V[S, NKV*HD] f32 -> Vt[NKV, HD, S] bf16 (B operand of P*V contiguous in keys)
// ---------------------------------------------------------------------------
__global__ void v_transpose_kernel(const float* __restrict__ V,
                                   __bf16* __restrict__ Vt) {
    int i = blockIdx.x * 256 + threadIdx.x;
    if (i >= NKV * HD * S_LEN) return;
    int s  = i % S_LEN;
    int t  = i / S_LEN;
    int d  = t % HD;
    int kv = t / HD;
    Vt[i] = (__bf16)V[(size_t)s * (NKV * HD) + kv * HD + d];
}

// ---------------------------------------------------------------------------
// Flash attention, GQA 16/4 heads. Block = 1 head x 128 queries (8 waves x 16),
// 32 keys/iter. K (32x128) and Vt (128x32) tiles double-buffered in LDS via
// async-to-LDS copies overlapping the WMMAs. Qb pre-scaled by 1/sqrt(HD).
// ---------------------------------------------------------------------------
__global__ __launch_bounds__(256) void flash_attn_kernel(
    const __bf16* __restrict__ Qb, const __bf16* __restrict__ Kb,
    const __bf16* __restrict__ Vt, __bf16* __restrict__ Ob) {
    __shared__ __attribute__((aligned(16))) __bf16 Ks[2][32][136];
    __shared__ __attribute__((aligned(16))) __bf16 Vs[2][128][40];
    __shared__ __attribute__((aligned(16))) __bf16 Ps[8][16][40];

    const int tid  = threadIdx.x;
    const int wave = tid >> 5;
    const int lane = tid & 31;
    const int half = lane >> 4;
    const int l16  = lane & 15;
    const int h  = blockIdx.y;
    const int kv = h / (NH / NKV);
    const int q0 = blockIdx.x * 128 + wave * 16;

    const __bf16* Qh = Qb + (size_t)h  * S_LEN * HD;
    const __bf16* Kh = Kb + (size_t)kv * S_LEN * HD;
    const __bf16* Vh = Vt + (size_t)kv * HD * S_LEN;

    auto stage = [&](int kc, int buf) {
        #pragma unroll
        for (int c = 0; c < 2; ++c) {
            int chunk = tid * 2 + c;
            {
                int row = chunk >> 4, col = (chunk & 15) * 8;
                async_copy_b128(&Ks[buf][row][col],
                                &Kh[(size_t)(kc + row) * HD + col]);
            }
            {
                int row = chunk >> 2, col = (chunk & 3) * 8;
                async_copy_b128(&Vs[buf][row][col],
                                &Vh[(size_t)row * S_LEN + kc + col]);
            }
        }
    };

    // preload Q A-operands (16 rows x 128 dims = 4 K-chunks), held in VGPRs
    v16bf qa[4];
    #pragma unroll
    for (int kb = 0; kb < 4; ++kb) {
        const __bf16* qr = Qh + (size_t)(q0 + l16) * HD + kb * 32;
        qa[kb] = cat8(*(const bf16x8*)&qr[half * 8],
                      *(const bf16x8*)&qr[16 + half * 8]);
    }

    float Mrow[8], Lrow[8];
    v8f o[8] = {};
    #pragma unroll
    for (int r = 0; r < 8; ++r) { Mrow[r] = -3.0e38f; Lrow[r] = 0.0f; }

    stage(0, 0);
    async_wait0();
    __syncthreads();

    for (int kc = 0, it = 0; kc < S_LEN; kc += 32, ++it) {
        const int buf = it & 1;
        if (kc + 32 < S_LEN) stage(kc + 32, buf ^ 1);   // overlaps WMMAs below

        // scores: two 16-key tiles, 4 K-dim WMMA steps each (operands from LDS)
        v8f s0 = {}, s1 = {};
        #pragma unroll
        for (int kb = 0; kb < 4; ++kb) {
            const __bf16* b0p = &Ks[buf][l16][kb * 32];
            const __bf16* b1p = &Ks[buf][16 + l16][kb * 32];
            s0 = wmma_bf16(qa[kb], cat8(*(const bf16x8*)&b0p[half * 8], *(const bf16x8*)&b0p[16 + half * 8]), s0);
            s1 = wmma_bf16(qa[kb], cat8(*(const bf16x8*)&b1p[half * 8], *(const bf16x8*)&b1p[16 + half * 8]), s1);
        }

        // online softmax: row stats via VALU permlane16 xor-reductions
        float al[8], p0[8], p1[8];
        #pragma unroll
        for (int r = 0; r < 8; ++r) {
            float v = red_max16(fmaxf(s0[r], s1[r]));
            float mn = fmaxf(Mrow[r], v);
            al[r] = __expf(Mrow[r] - mn);
            Mrow[r] = mn;
            p0[r] = __expf(s0[r] - mn);
            p1[r] = __expf(s1[r] - mn);
            float rs = red_sum16(p0[r] + p1[r]);
            Lrow[r] = Lrow[r] * al[r] + rs;
        }
        #pragma unroll
        for (int d = 0; d < 8; ++d)
            #pragma unroll
            for (int r = 0; r < 8; ++r)
                o[d][r] *= al[r];

        // C-layout -> A-layout via private per-wave LDS tile (same-wave LDS is
        // in-order; cross-wave sync not required for a wave-private buffer)
        #pragma unroll
        for (int r = 0; r < 8; ++r) {
            Ps[wave][half * 8 + r][l16]      = (__bf16)p0[r];
            Ps[wave][half * 8 + r][16 + l16] = (__bf16)p1[r];
        }
        bf16x8 plo = *(const bf16x8*)&Ps[wave][l16][half * 8];
        bf16x8 phi = *(const bf16x8*)&Ps[wave][l16][16 + half * 8];
        v16bf pa = cat8(plo, phi);

        // P(16x32) * V(32x128): 8 dim-chunks, B operand from LDS Vs
        #pragma unroll
        for (int db = 0; db < 8; ++db) {
            const __bf16* vp = &Vs[buf][db * 16 + l16][0];
            o[db] = wmma_bf16(pa, cat8(*(const bf16x8*)&vp[half * 8], *(const bf16x8*)&vp[16 + half * 8]), o[db]);
        }
        if (kc + 32 < S_LEN) async_wait0();
        __syncthreads();   // tiles for next iteration now complete & visible
    }

    #pragma unroll
    for (int r = 0; r < 8; ++r) {
        float inv = 1.0f / Lrow[r];
        int srow = q0 + half * 8 + r;
        #pragma unroll
        for (int db = 0; db < 8; ++db)
            Ob[(size_t)srow * (NH * HD) + h * HD + db * 16 + l16] =
                (__bf16)(o[db][r] * inv);
    }
}

// ---------------------------------------------------------------------------
extern "C" void kernel_launch(void* const* d_in, const int* in_sizes, int n_in,
                              void* d_out, int out_size, void* d_ws, size_t ws_size,
                              hipStream_t stream) {
    (void)in_sizes; (void)n_in; (void)out_size; (void)ws_size;
    const float* hs = (const float*)d_in[0];
    const float* Wq = (const float*)d_in[1];
    const float* Wk = (const float*)d_in[2];
    const float* Wv = (const float*)d_in[3];
    const float* Wo = (const float*)d_in[4];
    float* out = (float*)d_out;

    char* ws = (char*)d_ws;
    size_t off = 0;
    auto carve = [&](size_t bytes) {
        char* p = ws + off;
        off += (bytes + 255) & ~(size_t)255;
        return p;
    };
    __bf16* hsb = (__bf16*)carve((size_t)S_LEN * DMODEL * 2);
    __bf16* WqT = (__bf16*)carve((size_t)DMODEL * (NH * HD) * 2);
    __bf16* WkT = (__bf16*)carve((size_t)DMODEL * (NKV * HD) * 2);
    __bf16* WvT = (__bf16*)carve((size_t)DMODEL * (NKV * HD) * 2);
    __bf16* WoT = (__bf16*)carve((size_t)(NH * HD) * DMODEL * 2);
    float*  Qf  = (float*)carve((size_t)S_LEN * (NH * HD) * 4);
    float*  Kf  = (float*)carve((size_t)S_LEN * (NKV * HD) * 4);
    float*  Vf  = (float*)carve((size_t)S_LEN * (NKV * HD) * 4);
    __bf16* Qb  = (__bf16*)carve((size_t)NH * S_LEN * HD * 2);
    __bf16* Kb  = (__bf16*)carve((size_t)NKV * S_LEN * HD * 2);
    __bf16* Vt  = (__bf16*)carve((size_t)NKV * HD * S_LEN * 2);
    __bf16* Ob  = (__bf16*)carve((size_t)S_LEN * (NH * HD) * 2);

    const int T = 256;
    // stage bf16 operands
    convert_bf16_kernel<<<(S_LEN * DMODEL + T - 1) / T, T, 0, stream>>>(hs, hsb, S_LEN * DMODEL);
    transpose_bf16_kernel<<<(DMODEL * NH * HD + T - 1) / T, T, 0, stream>>>(Wq, WqT, DMODEL, NH * HD);
    transpose_bf16_kernel<<<(DMODEL * NKV * HD + T - 1) / T, T, 0, stream>>>(Wk, WkT, DMODEL, NKV * HD);
    transpose_bf16_kernel<<<(DMODEL * NKV * HD + T - 1) / T, T, 0, stream>>>(Wv, WvT, DMODEL, NKV * HD);
    transpose_bf16_kernel<<<(NH * HD * DMODEL + T - 1) / T, T, 0, stream>>>(Wo, WoT, NH * HD, DMODEL);

    // QKV projections (bf16 WMMA GEMM, f32 out). Block tile 128x128.
    gemm_bf16_kernel<S_LEN, NH * HD, DMODEL>
        <<<dim3((NH * HD) / 128, S_LEN / 128), T, 0, stream>>>(hsb, WqT, Qf);
    gemm_bf16_kernel<S_LEN, NKV * HD, DMODEL>
        <<<dim3((NKV * HD) / 128, S_LEN / 128), T, 0, stream>>>(hsb, WkT, Kf);
    gemm_bf16_kernel<S_LEN, NKV * HD, DMODEL>
        <<<dim3((NKV * HD) / 128, S_LEN / 128), T, 0, stream>>>(hsb, WvT, Vf);

    // RoPE + layout (fold 1/sqrt(HD) into Q), V transpose
    const float qscale = 0.08838834764831845f; // 1/sqrt(128)
    rope_pack_kernel<<<(S_LEN * NH * (HD / 2) + T - 1) / T, T, 0, stream>>>(Qf, Qb, NH, qscale);
    rope_pack_kernel<<<(S_LEN * NKV * (HD / 2) + T - 1) / T, T, 0, stream>>>(Kf, Kb, NKV, 1.0f);
    v_transpose_kernel<<<(NKV * HD * S_LEN + T - 1) / T, T, 0, stream>>>(Vf, Vt);

    // flash attention
    flash_attn_kernel<<<dim3(S_LEN / 128, NH), T, 0, stream>>>(Qb, Kb, Vt, Ob);

    // output projection straight into d_out (f32)
    gemm_bf16_kernel<S_LEN, DMODEL, NH * HD>
        <<<dim3(DMODEL / 128, S_LEN / 128), T, 0, stream>>>(Ob, WoT, out);
}
